// G2TotalLoss_10247791968549
// MI455X (gfx1250) — matrix operand order repaired and checked by reading, not compiled, not run on hardware
//
#include <hip/hip_runtime.h>
#include <math.h>

// ---------- types ----------
typedef __attribute__((ext_vector_type(16))) _Float16 v16h;
typedef __attribute__((ext_vector_type(8)))  _Float16 v8h;
typedef __attribute__((ext_vector_type(8)))  float    v8f;

#define B_TOTAL 262144
#define DD 7
#define HH 128
#define NCOMP 35
#define NPAD 48               // phi columns padded to 3 WMMA tiles
#define KPAD 64               // phi K-dim padded to 2 WMMA chunks
#define WAVES_PER_BLOCK 2
#define SQRT7F 2.6457513110645906f

// fast tanh on the v_exp_f32 path: tanh(z) = 1 - 2/(e^{2z}+1)
__device__ __forceinline__ float fast_tanh(float z) {
  return 1.f - 2.f / (__expf(2.f * z) + 1.f);
}

// ============================================================
// Precompute (once, tiny):
//   G1[h,h'] = sum_j W1[j,h] W1[j,h']            (f16 128x128)
//   Gc[h,h'] = G1[h,h'] * sum_k W2[h,k] W2[h',k] (f16 128x128)
//   w2t[n][k]   = W2[k,n]  (f16 48x128, zero-padded)  B for phi-GEMM
//   w2r64[h][k] = W2[h,k]  (f16 128x64, zero-padded)  B for v-GEMM
// ============================================================
__global__ void g2_precompute_gram(const float* __restrict__ W1,
                                   const float* __restrict__ W2,
                                   _Float16* __restrict__ gc,
                                   _Float16* __restrict__ g1o,
                                   _Float16* __restrict__ w2t,
                                   _Float16* __restrict__ w2r64) {
  const int h  = blockIdx.x;    // 0..127
  const int hp = threadIdx.x;   // 0..127
  float g1 = 0.f;
  #pragma unroll
  for (int j = 0; j < DD; ++j) g1 += W1[j * HH + h] * W1[j * HH + hp];
  float g2 = 0.f;
  #pragma unroll
  for (int k = 0; k < NCOMP; ++k) g2 += W2[h * NCOMP + k] * W2[hp * NCOMP + k];
  g1o[h * HH + hp] = (_Float16)g1;
  gc [h * HH + hp] = (_Float16)(g1 * g2);
  if (hp < KPAD)
    w2r64[h * KPAD + hp] = (_Float16)(hp < NCOMP ? W2[h * NCOMP + hp] : 0.f);
  if (h < NPAD)
    w2t[h * HH + hp] = (_Float16)(h < NCOMP ? W2[hp * NCOMP + h] : 0.f);
}

__global__ void g2_zero_out(float* out) { out[0] = 0.f; }

// ---------- WMMA fragment helpers ----------
// A-fragment (16x32 f16): lane (m,hi) holds row m; slot p needs
// k = 32*kc + (p<8 ? p : 8+p) + 8*hi -> two contiguous 16B chunks of the
// lane's own staged row (which holds exactly the k with (k>>3)&1 == hi).
__device__ __forceinline__ v16h load_afrag(const _Float16* rowBase, int kc, int hi) {
  const v8h* p = (const v8h*)rowBase + 4 * kc + hi;
  v8h lo = p[0];   // k = 32kc+8hi + 0..7
  v8h up = p[2];   // k = 32kc+8hi + 16..23
  v16h r;
  #pragma unroll
  for (int i = 0; i < 8; ++i) { r[i] = lo[i]; r[i + 8] = up[i]; }
  return r;
}

// B-fragment (32x16 f16): lane holds column n; slot p holds
// K = (p<8 ? p : 8+p) + 8*hi of the chunk. Storage is row-major [n][k] with 'stride'.
__device__ __forceinline__ v16h load_bfrag(const _Float16* __restrict__ G,
                                           int n, int kc, int hi, int stride) {
  const v8h* p = (const v8h*)(G + n * stride + 32 * kc + 8 * hi);
  v8h a = p[0], b = p[2];
  v16h r;
  #pragma unroll
  for (int i = 0; i < 8; ++i) { r[i] = a[i]; r[i + 8] = b[i]; }
  return r;
}

__device__ __forceinline__ v8f wmma16(v16h a, v16h b, v8f c) {
  return __builtin_amdgcn_wmma_f32_16x16x32_f16(false, a, false, b,
                                                (short)0, c, false, false);
}

// ============================================================
// Main kernel: 64 threads = 2 waves; each wave owns a tile of 16 samples.
// All four per-sample matmuls (phi, v, s^T G s, u^T G1 u) run on WMMA.
// ============================================================
__launch_bounds__(64)
__global__ void g2_main(const float* __restrict__ coords,
                        const float* __restrict__ metric,
                        const float* __restrict__ W1g,
                        const float* __restrict__ b1g,
                        const float* __restrict__ b2g,
                        const _Float16* __restrict__ gc,
                        const _Float16* __restrict__ g1m,
                        const _Float16* __restrict__ w2t,
                        const _Float16* __restrict__ w2r64,
                        float* __restrict__ out) {
  __shared__ float w1s[DD * HH];
  __shared__ float b1s[HH];
  __shared__ float b2s[NPAD];
  __shared__ float    __align__(16) Rs [WAVES_PER_BLOCK][16 * HH];   // f32 D-tiles
  __shared__ _Float16 __align__(16) Ss [WAVES_PER_BLOCK][16 * HH];   // s (later u)
  __shared__ _Float16 __align__(16) Ts [WAVES_PER_BLOCK][16 * HH];   // tanh(z)
  __shared__ float    __align__(16) Ps [WAVES_PER_BLOCK][16 * NPAD]; // phi f32
  __shared__ _Float16 __align__(16) Psh[WAVES_PER_BLOCK][16 * KPAD]; // phi f16 A

  const int tid  = threadIdx.x;
  const int wave = tid >> 5;
  const int lane = tid & 31;
  const int m    = lane & 15;
  const int hi   = lane >> 4;   // which k-half this lane owns: (k>>3)&1 == hi

  // warm L2 with the shared f16 operand tables (global_prefetch_b8)
  __builtin_prefetch((const void*)(gc  + tid * 256), 0, 3);
  __builtin_prefetch((const void*)(g1m + tid * 256), 0, 3);
  __builtin_prefetch((const void*)(w2t + tid * 96), 0, 3);
  __builtin_prefetch((const void*)(w2r64 + tid * 128), 0, 3);

  // ---- stage small weights in LDS ----
  for (int i = tid; i < DD * HH; i += 64) w1s[i] = W1g[i];
  for (int i = tid; i < HH; i += 64)      b1s[i] = b1g[i];
  for (int i = tid; i < NPAD; i += 64)    b2s[i] = (i < NCOMP) ? b2g[i] : 0.f;

  const int sample = (blockIdx.x * WAVES_PER_BLOCK + wave) * 16 + m;
  const float* cp = coords + (size_t)sample * DD;
  const float c0 = cp[0], c1 = cp[1], c2 = cp[2], c3 = cp[3],
              c4 = cp[4], c5 = cp[5], c6 = cp[6];

  float*     RsW    = Rs[wave];
  float*     PsW    = Ps[wave];
  _Float16*  PshW   = Psh[wave];
  _Float16*  SsRow  = Ss[wave] + m * HH;
  _Float16*  TsRow  = Ts[wave] + m * HH;
  _Float16*  PshRow = PshW + m * KPAD;

  // zero the phi K-padding region (cols 48..63) of this wave's Psh
  for (int i = lane; i < 16 * 16; i += 32)
    PshW[(i >> 4) * KPAD + NPAD + (i & 15)] = (_Float16)0.f;
  __syncthreads();

  // ---- phase 1: z = cW1+b1, t = tanh(z), s = 1-t^2 ----
  for (int q = 0; q < 64; ++q) {
    const int h = ((q >> 3) << 4) | (hi << 3) | (q & 7);
    const float* w1c = w1s + h;
    float z = b1s[h] + c0 * w1c[0]      + c1 * w1c[HH]     + c2 * w1c[2 * HH]
                     + c3 * w1c[3 * HH] + c4 * w1c[4 * HH] + c5 * w1c[5 * HH]
                     + c6 * w1c[6 * HH];
    float t = fast_tanh(z);
    TsRow[h] = (_Float16)t;
    SsRow[h] = (_Float16)(1.f - t * t);
  }
  __syncthreads();

  // ---- phi-GEMM: phi = T @ W2 + b2   (16x128 @ 128x48) : 12 WMMAs ----
  {
    v16h tf0 = load_afrag(TsRow, 0, hi);
    v16h tf1 = load_afrag(TsRow, 1, hi);
    v16h tf2 = load_afrag(TsRow, 2, hi);
    v16h tf3 = load_afrag(TsRow, 3, hi);
    #pragma unroll
    for (int nt = 0; nt < 3; ++nt) {
      const int n = 16 * nt + m;
      const float bb = b2s[n];
      v8f acc = {bb, bb, bb, bb, bb, bb, bb, bb};   // bias as C operand
      acc = wmma16(tf0, load_bfrag(w2t, n, 0, hi, HH), acc);
      acc = wmma16(tf1, load_bfrag(w2t, n, 1, hi, HH), acc);
      acc = wmma16(tf2, load_bfrag(w2t, n, 2, hi, HH), acc);
      acc = wmma16(tf3, load_bfrag(w2t, n, 3, hi, HH), acc);
      #pragma unroll
      for (int r = 0; r < 8; ++r) {
        const int M = r + 8 * hi;                    // D layout
        PsW [M * NPAD + n] = acc[r];
        PshW[M * KPAD + n] = (_Float16)acc[r];
      }
    }
  }
  __syncthreads();

  // ---- |phi|^2 ----
  float pn2 = 0.f;
  {
    const float* prow = PsW + m * NPAD;
    #pragma unroll
    for (int k = 0; k < NCOMP; ++k) pn2 += prow[k] * prow[k];
  }

  // ---- S-GEMM: Y = S @ (G1 o G2) : 32 WMMAs ----
  {
    v16h af0 = load_afrag(SsRow, 0, hi);
    v16h af1 = load_afrag(SsRow, 1, hi);
    v16h af2 = load_afrag(SsRow, 2, hi);
    v16h af3 = load_afrag(SsRow, 3, hi);
    #pragma unroll
    for (int nt = 0; nt < 8; ++nt) {
      const int n = 16 * nt + m;
      v8f acc = {0.f, 0.f, 0.f, 0.f, 0.f, 0.f, 0.f, 0.f};
      acc = wmma16(af0, load_bfrag(gc, n, 0, hi, HH), acc);
      acc = wmma16(af1, load_bfrag(gc, n, 1, hi, HH), acc);
      acc = wmma16(af2, load_bfrag(gc, n, 2, hi, HH), acc);
      acc = wmma16(af3, load_bfrag(gc, n, 3, hi, HH), acc);
      #pragma unroll
      for (int r = 0; r < 8; ++r) RsW[(r + 8 * hi) * HH + n] = acc[r];
    }
  }
  __syncthreads();

  // ---- dphi = s^T G s = sum_h Y[m,h] * s[m,h] ----
  float dphi = 0.f;
  #pragma unroll
  for (int g = 0; g < 8; ++g) {
    const int h0 = 16 * g + 8 * hi;
    const v8h sv = *(const v8h*)(SsRow + h0);
    const float* rr = RsW + m * HH + h0;
    #pragma unroll
    for (int p = 0; p < 8; ++p) dphi += (float)sv[p] * rr[p];
  }
  dphi += __shfl_xor(dphi, 16, 32);
  __syncthreads();

  // ---- v-GEMM: v = phi @ W2^T  (16x64 @ 64x128) : 16 WMMAs -> Rs ----
  {
    v16h pf0 = load_afrag(PshRow, 0, hi);
    v16h pf1 = load_afrag(PshRow, 1, hi);
    #pragma unroll
    for (int nt = 0; nt < 8; ++nt) {
      const int n = 16 * nt + m;
      v8f acc = {0.f, 0.f, 0.f, 0.f, 0.f, 0.f, 0.f, 0.f};
      acc = wmma16(pf0, load_bfrag(w2r64, n, 0, hi, KPAD), acc);
      acc = wmma16(pf1, load_bfrag(w2r64, n, 1, hi, KPAD), acc);
      #pragma unroll
      for (int r = 0; r < 8; ++r) RsW[(r + 8 * hi) * HH + n] = acc[r];
    }
  }
  __syncthreads();

  // ---- u = s * v -> overwrite Ss (same-lane data only) ----
  for (int q = 0; q < 64; ++q) {
    const int h = ((q >> 3) << 4) | (hi << 3) | (q & 7);
    SsRow[h] = (_Float16)((float)SsRow[h] * RsW[m * HH + h]);
  }
  __syncthreads();

  // ---- U-GEMM: Q = U @ G1 : 32 WMMAs ----
  {
    v16h af0 = load_afrag(SsRow, 0, hi);
    v16h af1 = load_afrag(SsRow, 1, hi);
    v16h af2 = load_afrag(SsRow, 2, hi);
    v16h af3 = load_afrag(SsRow, 3, hi);
    #pragma unroll
    for (int nt = 0; nt < 8; ++nt) {
      const int n = 16 * nt + m;
      v8f acc = {0.f, 0.f, 0.f, 0.f, 0.f, 0.f, 0.f, 0.f};
      acc = wmma16(af0, load_bfrag(g1m, n, 0, hi, HH), acc);
      acc = wmma16(af1, load_bfrag(g1m, n, 1, hi, HH), acc);
      acc = wmma16(af2, load_bfrag(g1m, n, 2, hi, HH), acc);
      acc = wmma16(af3, load_bfrag(g1m, n, 3, hi, HH), acc);
      #pragma unroll
      for (int r = 0; r < 8; ++r) RsW[(r + 8 * hi) * HH + n] = acc[r];
    }
  }
  __syncthreads();

  // ---- qq = u^T G1 u ----
  float qq = 0.f;
  #pragma unroll
  for (int g = 0; g < 8; ++g) {
    const int h0 = 16 * g + 8 * hi;
    const v8h uv = *(const v8h*)(SsRow + h0);
    const float* rr = RsW + m * HH + h0;
    #pragma unroll
    for (int p = 0; p < 8; ++p) qq += (float)uv[p] * rr[p];
  }
  qq += __shfl_xor(qq, 16, 32);

  // ---- per-sample scalars ----
  // high lanes: exact det via Gaussian elimination.
  // low lanes: Gershgorin early-out for the positivity penalty; exact
  //            Jacobi eigensolve only for the (rare) samples that might
  //            have an eigenvalue below 1e-6.
  const float* Mp = metric + (size_t)sample * (DD * DD);
  float detv = 0.f, eigpen = 0.f;
  if (hi) {
    float a[DD * DD];
    for (int i = 0; i < DD * DD; ++i) a[i] = Mp[i];
    float det = 1.f;
    for (int kk = 0; kk < DD; ++kk) {
      const float piv = a[kk * DD + kk];
      det *= piv;
      const float ip = 1.f / piv;
      for (int r = kk + 1; r < DD; ++r) {
        const float f = a[r * DD + kk] * ip;
        for (int cc = kk; cc < DD; ++cc) a[r * DD + cc] -= f * a[kk * DD + cc];
      }
    }
    detv = det;
  } else {
    // Gershgorin lower bound: min_i (a_ii - sum_{j!=i} |a_ij|) <= min eigenvalue
    float minb = 1e30f;
    #pragma unroll
    for (int i = 0; i < DD; ++i) {
      float off = 0.f;
      #pragma unroll
      for (int j = 0; j < DD; ++j)
        if (j != i) off += fabsf(Mp[i * DD + j]);
      const float bnd = Mp[i * DD + i] - off;
      minb = fminf(minb, bnd);
    }
    if (minb <= 1e-6f) {
      // rare path: exact cyclic Jacobi eigensolve
      float a[DD * DD];
      for (int i = 0; i < DD * DD; ++i) a[i] = Mp[i];
      for (int sweep = 0; sweep < 6; ++sweep) {
        for (int p = 0; p < DD - 1; ++p) {
          for (int r = p + 1; r < DD; ++r) {
            const float apq = a[p * DD + r];
            if (fabsf(apq) > 1e-30f) {
              const float tau = (a[r * DD + r] - a[p * DD + p]) / (2.f * apq);
              const float tt  = (tau >= 0.f ? 1.f : -1.f) /
                                (fabsf(tau) + sqrtf(tau * tau + 1.f));
              const float cc  = 1.f / sqrtf(tt * tt + 1.f);
              const float sn  = tt * cc;
              for (int i = 0; i < DD; ++i) {
                const float xip = a[i * DD + p], xir = a[i * DD + r];
                a[i * DD + p] = cc * xip - sn * xir;
                a[i * DD + r] = sn * xip + cc * xir;
              }
              for (int i = 0; i < DD; ++i) {
                const float xpi = a[p * DD + i], xri = a[r * DD + i];
                a[p * DD + i] = cc * xpi - sn * xri;
                a[r * DD + i] = sn * xpi + cc * xri;
              }
            }
          }
        }
      }
      for (int i = 0; i < DD; ++i) {
        const float rl = fmaxf(1e-6f - a[i * DD + i], 0.f);
        eigpen += rl * rl;
      }
    }
    // else: all eigenvalues provably > 1e-6 -> penalty is exactly 0
  }
  const float det = __shfl_xor(detv, 16, 32);  // low lanes receive exact det

  // ---- per-sample loss combine (valid on low lanes), wave reduce ----
  float contrib;
  {
    const float vol  = sqrtf(fabsf(det) + 1e-10f);
    const float pn   = fmaxf(sqrtf(pn2), 1e-20f);
    const float nden = vol * pn + 1e-8f;
    const float aco  = SQRT7F * vol / nden;
    const float bco  = vol / (pn * nden);
    const float ddual = aco * aco * (dphi - (2.f * bco - bco * bco * pn2) * qq);
    const float lt = dphi + ddual;                 // torsion (W=1)
    const float lv = (det - 1.f) * (det - 1.f);    // volume  (W=1)
    const float dn = pn2 - 7.f;
    const float ln = dn * dn;                      // phi-norm
    contrib = (lt + lv + ln + eigpen * (1.f / 7.f)) * (1.f / (float)B_TOTAL);
  }
  contrib = hi ? 0.f : contrib;
  contrib += __shfl_xor(contrib, 1, 32);
  contrib += __shfl_xor(contrib, 2, 32);
  contrib += __shfl_xor(contrib, 4, 32);
  contrib += __shfl_xor(contrib, 8, 32);
  contrib += __shfl_xor(contrib, 16, 32);
  if (lane == 0) atomicAdd(out, contrib);
}

// ============================================================
extern "C" void kernel_launch(void* const* d_in, const int* in_sizes, int n_in,
                              void* d_out, int out_size, void* d_ws, size_t ws_size,
                              hipStream_t stream) {
  const float* coords = (const float*)d_in[0];  // [B,7]
  const float* metric = (const float*)d_in[1];  // [B,7,7]
  const float* W1     = (const float*)d_in[2];  // [7,128]
  const float* b1     = (const float*)d_in[3];  // [128]
  const float* W2     = (const float*)d_in[4];  // [128,35]
  const float* b2     = (const float*)d_in[5];  // [35]
  float* out = (float*)d_out;

  _Float16* gc    = (_Float16*)d_ws;        // (G1 o G2) f16 [128x128]
  _Float16* g1m   = gc  + HH * HH;          // G1 f16 [128x128]
  _Float16* w2t   = g1m + HH * HH;          // W2^T f16 [48x128] (zero-padded)
  _Float16* w2r64 = w2t + NPAD * HH;        // W2 f16 [128x64]   (zero-padded)

  g2_precompute_gram<<<dim3(HH), dim3(HH), 0, stream>>>(W1, W2, gc, g1m,
                                                        w2t, w2r64);
  g2_zero_out<<<dim3(1), dim3(1), 0, stream>>>(out);
  g2_main<<<dim3(B_TOTAL / (16 * WAVES_PER_BLOCK)), dim3(32 * WAVES_PER_BLOCK),
            0, stream>>>(coords, metric, W1, b1, b2, gc, g1m, w2t, w2r64, out);
}
